// MoE_50946902065666
// MI455X (gfx1250) — matrix operand order repaired and checked by reading, not compiled, not run on hardware
//
#include <hip/hip_runtime.h>
#include <hip/hip_bf16.h>

// ---------------- problem dims (B=4,S=2048,D=2048,E=8,I=1408,SI=2816) -------
constexpr int T_TOK = 8192;
constexpr int DIM   = 2048;
constexpr int NEXP  = 8;
constexpr int IDIM  = 1408;
constexpr int SIDIM = 2816;

constexpr int TILE_M = 128;
constexpr int TILE_N = 64;
constexpr int KSTEP  = 32;
constexpr int LDT    = KSTEP + 8;   // padded LDS row (bf16 elems)

typedef __attribute__((ext_vector_type(16))) __bf16 v16bf;
typedef __attribute__((ext_vector_type(8)))  float  v8f;

union FragU { uint4 q[2]; v16bf v; };

// copy 16 bytes global(bf16) -> LDS via CDNA5 async-LDS DMA (ASYNCcnt-tracked).
// VDST = wave-relative LDS byte offset (low 32 bits of generic shared pointer),
// VADDR = 64-bit global address, GV mode (saddr = off).
static __device__ __forceinline__ void cp16(const unsigned short* g, unsigned short* l) {
  unsigned lofs = (unsigned)(unsigned long long)l;   // LDS aperture: low 32 bits = offset
  asm volatile("global_load_async_to_lds_b128 %0, %1, off"
               :
               : "v"(lofs), "v"(g)
               : "memory");
}

static __device__ __forceinline__ void wait_async() {
#if __has_builtin(__builtin_amdgcn_s_wait_asynccnt)
  __builtin_amdgcn_s_wait_asynccnt(0);
#else
  asm volatile("s_wait_asynccnt 0x0" ::: "memory");
#endif
}

static __device__ __forceinline__ unsigned short f2bf(float f) {
  unsigned u = __float_as_uint(f);
  unsigned r = u + 0x7FFFu + ((u >> 16) & 1u);  // round-to-nearest-even
  return (unsigned short)(r >> 16);
}

static __device__ __forceinline__ v8f zero8() {
  v8f z;
#pragma unroll
  for (int q = 0; q < 8; ++q) z[q] = 0.f;
  return z;
}

// ---------------- fp32 -> bf16 streaming convert (x and all weights) --------
__global__ __launch_bounds__(256)
void cvt_bf16_kernel(const float* __restrict__ x, unsigned short* __restrict__ xb, int n4) {
  int i = blockIdx.x * 256 + threadIdx.x;
  if (i >= n4) return;
  float4 f = ((const float4*)x)[i];
  uint2 o;
  o.x = (unsigned)f2bf(f.x) | ((unsigned)f2bf(f.y) << 16);
  o.y = (unsigned)f2bf(f.z) | ((unsigned)f2bf(f.w) << 16);
  ((uint2*)xb)[i] = o;
}

// ---------------- gating: one wave per token --------------------------------
__global__ void zero_counts_kernel(int* counts) {
  if (threadIdx.x < NEXP) counts[threadIdx.x] = 0;
}

__global__ __launch_bounds__(256)
void gate_topk_kernel(const float* __restrict__ x, const float* __restrict__ gw,
                      int* __restrict__ topi, float* __restrict__ topw,
                      int* __restrict__ counts) {
  int wave = threadIdx.x >> 5, lane = threadIdx.x & 31;
  int t = blockIdx.x * 8 + wave;
  if (t >= T_TOK) return;
  const float* xr = x + (size_t)t * DIM;
  float acc[NEXP];
#pragma unroll
  for (int e = 0; e < NEXP; ++e) acc[e] = 0.f;
  for (int d = lane; d < DIM; d += 32) {
    float xv = xr[d];
#pragma unroll
    for (int e = 0; e < NEXP; ++e) acc[e] += xv * gw[e * DIM + d];
  }
#pragma unroll
  for (int e = 0; e < NEXP; ++e) {
    float v = acc[e];
    for (int off = 16; off; off >>= 1) v += __shfl_xor(v, off, 32);
    acc[e] = v;
  }
  if (lane == 0) {
    float mx = acc[0];
#pragma unroll
    for (int e = 1; e < NEXP; ++e) mx = fmaxf(mx, acc[e]);
    float p[NEXP], sum = 0.f;
#pragma unroll
    for (int e = 0; e < NEXP; ++e) { p[e] = __expf(acc[e] - mx); sum += p[e]; }
    float inv = 1.f / sum;
    int i1 = 0; float b1 = -1.f;
#pragma unroll
    for (int e = 0; e < NEXP; ++e) if (p[e] > b1) { b1 = p[e]; i1 = e; }
    int i2 = 0; float b2 = -1.f;
#pragma unroll
    for (int e = 0; e < NEXP; ++e) if (e != i1 && p[e] > b2) { b2 = p[e]; i2 = e; }
    topi[t * 2 + 0] = i1; topi[t * 2 + 1] = i2;
    topw[t * 2 + 0] = b1 * inv; topw[t * 2 + 1] = b2 * inv;  // ROUTE_SCALE=1
    atomicAdd(&counts[i1], 1);
    atomicAdd(&counts[i2], 1);
  }
}

__global__ void prefix8_kernel(const int* counts, int* segOff, int* cursor) {
  if (threadIdx.x == 0) {
    int s = 0;
    for (int e = 0; e < NEXP; ++e) { segOff[e] = s; cursor[e] = s; s += counts[e]; }
    segOff[NEXP] = s;
  }
}

__global__ __launch_bounds__(256)
void scatter_rows_kernel(const int* __restrict__ topi, const float* __restrict__ topw,
                         int* __restrict__ cursor, int* __restrict__ rowsA,
                         float* __restrict__ rowWA) {
  int t = blockIdx.x * 256 + threadIdx.x;
  if (t >= T_TOK) return;
#pragma unroll
  for (int k = 0; k < 2; ++k) {
    int e = topi[t * 2 + k];
    int p = atomicAdd(&cursor[e], 1);
    rowsA[p] = t;
    rowWA[p] = topw[t * 2 + k];
  }
}

// ---------------- SwiGLU up-projection GEMM (dual accumulator) --------------
// h[r, n] = silu(x[r]·W1[n]) * (x[r]·W3[n]);  A bf16 [rows,K], W bf16 [N,K]
template<bool GATHER>
__global__ __launch_bounds__(256)
void up_swiglu_kernel(const unsigned short* __restrict__ xb,
                      const unsigned short* __restrict__ w1g,
                      const unsigned short* __restrict__ w3g,
                      unsigned short* __restrict__ hout,
                      const int* __restrict__ rows, const int* __restrict__ segOff,
                      int Ncols, int Kdim, int denseRows) {
  __shared__ unsigned short As[TILE_M][LDT];
  __shared__ unsigned short Bs[2][TILE_N][LDT];

  int rowBase = 0, rowCount = denseRows;
  const unsigned short* w1 = w1g;
  const unsigned short* w3 = w3g;
  if (GATHER) {
    int e = blockIdx.z;
    rowBase = segOff[e];
    rowCount = segOff[e + 1] - rowBase;
    size_t wsz = (size_t)Ncols * Kdim;
    w1 = w1g + (size_t)e * wsz;
    w3 = w3g + (size_t)e * wsz;
  }
  int m0 = blockIdx.y * TILE_M;
  if (m0 >= rowCount) return;
  int n0 = blockIdx.x * TILE_N;

  int tid = threadIdx.x;
  // A loader: thread -> (row, 16-elem segment); clamp OOB rows (never stored)
  int ar = tid >> 1;
  int aseg = (tid & 1) << 4;
  int arow = (m0 + ar < rowCount) ? (m0 + ar) : (rowCount - 1);
  int atok = GATHER ? rows[rowBase + arow] : arow;
  const unsigned short* asrc = xb + (size_t)atok * Kdim + aseg;
  unsigned short* adst = &As[ar][aseg];
  // B loader: thread -> (weight, n row, 16-elem segment)
  int wsel = tid >> 7;
  int bidx = tid & 127;
  int bn = bidx >> 1;
  int bseg = (bidx & 1) << 4;
  const unsigned short* bsrc = (wsel ? w3 : w1) + (size_t)(n0 + bn) * Kdim + bseg;
  unsigned short* bdst = &Bs[wsel][bn][bseg];

  int wave = tid >> 5, lane = tid & 31;
  int wm = (wave >> 1) << 5;   // 0,32,64,96
  int wn = (wave & 1) << 5;    // 0,32
  int halfl = lane >> 4, l16 = lane & 15;

  v8f acc1[2][2], acc3[2][2];
#pragma unroll
  for (int i = 0; i < 2; ++i)
#pragma unroll
    for (int j = 0; j < 2; ++j) { acc1[i][j] = zero8(); acc3[i][j] = zero8(); }

  for (int k0 = 0; k0 < Kdim; k0 += KSTEP) {
    __syncthreads();                       // previous iteration's reads done
    cp16(asrc + k0, adst);
    cp16(asrc + k0 + 8, adst + 8);
    cp16(bsrc + k0, bdst);
    cp16(bsrc + k0 + 8, bdst + 8);
    if (k0 + KSTEP < Kdim) {
      __builtin_prefetch(asrc + k0 + KSTEP, 0, 1);
      __builtin_prefetch(bsrc + k0 + KSTEP, 0, 1);
    }
    wait_async();
    __syncthreads();                       // tiles visible to all waves

    FragU af[2];
#pragma unroll
    for (int i = 0; i < 2; ++i) {
      const unsigned short* ap = &As[wm + i * 16 + l16][0];
      af[i].q[0] = *(const uint4*)(ap + halfl * 8);        // K 0..7 / 8..15
      af[i].q[1] = *(const uint4*)(ap + 16 + halfl * 8);   // K 16..23 / 24..31
    }
    FragU bf1[2], bf3[2];
#pragma unroll
    for (int j = 0; j < 2; ++j) {
      const unsigned short* bp = &Bs[0][wn + j * 16 + l16][0];
      bf1[j].q[0] = *(const uint4*)(bp + halfl * 8);
      bf1[j].q[1] = *(const uint4*)(bp + 16 + halfl * 8);
      bp = &Bs[1][wn + j * 16 + l16][0];
      bf3[j].q[0] = *(const uint4*)(bp + halfl * 8);
      bf3[j].q[1] = *(const uint4*)(bp + 16 + halfl * 8);
    }
#pragma unroll
    for (int i = 0; i < 2; ++i)
#pragma unroll
      for (int j = 0; j < 2; ++j) {
        acc1[i][j] = __builtin_amdgcn_wmma_f32_16x16x32_bf16(
            false, af[i].v, false, bf1[j].v, (short)0, acc1[i][j], false, false);
        acc3[i][j] = __builtin_amdgcn_wmma_f32_16x16x32_bf16(
            false, af[i].v, false, bf3[j].v, (short)0, acc3[i][j], false, false);
      }
  }

  // epilogue: h = silu(acc1) * acc3 -> bf16
#pragma unroll
  for (int i = 0; i < 2; ++i)
#pragma unroll
    for (int j = 0; j < 2; ++j) {
      int ncol = n0 + wn + j * 16 + l16;
#pragma unroll
      for (int v = 0; v < 8; ++v) {
        int grow = m0 + wm + i * 16 + (halfl << 3) + v;
        if (grow < rowCount) {
          float g = acc1[i][j][v];
          float h = (g / (1.f + __expf(-g))) * acc3[i][j][v];
          hout[(size_t)(rowBase + grow) * Ncols + ncol] = f2bf(h);
        }
      }
    }
}

// ---------------- down-projection GEMM --------------------------------------
// out[t, n] (+)= scale_r * (h[r]·W2[n]);  h bf16 [rows,K], W2 bf16 [N,K]
template<bool GATHER>
__global__ __launch_bounds__(256)
void down_proj_kernel(const unsigned short* __restrict__ hin,
                      const unsigned short* __restrict__ w2g, float* __restrict__ out,
                      const int* __restrict__ rows, const float* __restrict__ rowW,
                      const int* __restrict__ segOff,
                      int Ncols, int Kdim, int denseRows) {
  __shared__ unsigned short As[TILE_M][LDT];
  __shared__ unsigned short Bs[TILE_N][LDT];

  int rowBase = 0, rowCount = denseRows;
  const unsigned short* wB = w2g;
  if (GATHER) {
    int e = blockIdx.z;
    rowBase = segOff[e];
    rowCount = segOff[e + 1] - rowBase;
    wB = w2g + (size_t)e * Ncols * Kdim;
  }
  int m0 = blockIdx.y * TILE_M;
  if (m0 >= rowCount) return;
  int n0 = blockIdx.x * TILE_N;

  int tid = threadIdx.x;
  int ar = tid >> 1, aseg = (tid & 1) << 4;
  int arow = (m0 + ar < rowCount) ? (m0 + ar) : (rowCount - 1);
  const unsigned short* asrc = hin + (size_t)(rowBase + arow) * Kdim + aseg;
  unsigned short* adst = &As[ar][aseg];
  // B loader: 8-elem (16B) chunk per thread: 64 rows x 32 elems
  int bn = tid >> 2, bseg = (tid & 3) << 3;
  const unsigned short* bsrc = wB + (size_t)(n0 + bn) * Kdim + bseg;
  unsigned short* bdst = &Bs[bn][bseg];

  int wave = tid >> 5, lane = tid & 31;
  int wm = (wave >> 1) << 5, wn = (wave & 1) << 5;
  int halfl = lane >> 4, l16 = lane & 15;

  v8f acc[2][2];
#pragma unroll
  for (int i = 0; i < 2; ++i)
#pragma unroll
    for (int j = 0; j < 2; ++j) acc[i][j] = zero8();

  for (int k0 = 0; k0 < Kdim; k0 += KSTEP) {
    __syncthreads();
    cp16(asrc + k0, adst);
    cp16(asrc + k0 + 8, adst + 8);
    cp16(bsrc + k0, bdst);
    if (k0 + KSTEP < Kdim) {
      __builtin_prefetch(asrc + k0 + KSTEP, 0, 1);
      __builtin_prefetch(bsrc + k0 + KSTEP, 0, 1);
    }
    wait_async();
    __syncthreads();

    FragU af[2], bfg[2];
#pragma unroll
    for (int i = 0; i < 2; ++i) {
      const unsigned short* ap = &As[wm + i * 16 + l16][0];
      af[i].q[0] = *(const uint4*)(ap + halfl * 8);
      af[i].q[1] = *(const uint4*)(ap + 16 + halfl * 8);
    }
#pragma unroll
    for (int j = 0; j < 2; ++j) {
      const unsigned short* bp = &Bs[wn + j * 16 + l16][0];
      bfg[j].q[0] = *(const uint4*)(bp + halfl * 8);
      bfg[j].q[1] = *(const uint4*)(bp + 16 + halfl * 8);
    }
#pragma unroll
    for (int i = 0; i < 2; ++i)
#pragma unroll
      for (int j = 0; j < 2; ++j)
        acc[i][j] = __builtin_amdgcn_wmma_f32_16x16x32_bf16(
            false, af[i].v, false, bfg[j].v, (short)0, acc[i][j], false, false);
  }

#pragma unroll
  for (int i = 0; i < 2; ++i)
#pragma unroll
    for (int v = 0; v < 8; ++v) {
      int grow = m0 + wm + i * 16 + (halfl << 3) + v;
      if (grow >= rowCount) continue;
      int t; float s;
      if (GATHER) { t = rows[rowBase + grow]; s = rowW[rowBase + grow]; }
      else        { t = grow; s = 1.f; }
#pragma unroll
      for (int j = 0; j < 2; ++j) {
        int ncol = n0 + wn + j * 16 + l16;
        float val = acc[i][j][v] * s;
        if (GATHER) {
          (void)__hip_atomic_fetch_add(&out[(size_t)t * Ncols + ncol], val,
                                       __ATOMIC_RELAXED, __HIP_MEMORY_SCOPE_AGENT);
        } else {
          out[(size_t)t * Ncols + ncol] = val;
        }
      }
    }
}

// ---------------- launcher ---------------------------------------------------
extern "C" void kernel_launch(void* const* d_in, const int* in_sizes, int n_in,
                              void* d_out, int out_size, void* d_ws, size_t ws_size,
                              hipStream_t stream) {
  (void)in_sizes; (void)n_in; (void)out_size; (void)ws_size;
  const float* x   = (const float*)d_in[0];
  const float* gw  = (const float*)d_in[1];
  const float* w1  = (const float*)d_in[2];
  const float* w2  = (const float*)d_in[3];
  const float* w3  = (const float*)d_in[4];
  const float* sw1 = (const float*)d_in[5];
  const float* sw2 = (const float*)d_in[6];
  const float* sw3 = (const float*)d_in[7];
  float* out = (float*)d_out;

  char* ws = (char*)d_ws;
  size_t off = 0;
  auto take = [&](size_t bytes) -> char* {
    char* p = ws + off;
    off += (bytes + 255) & ~(size_t)255;
    return p;
  };
  const size_t WREXP = (size_t)NEXP * IDIM * DIM;   // per routed weight tensor
  const size_t WRSH  = (size_t)SIDIM * DIM;         // per shared weight tensor
  unsigned short* xb   = (unsigned short*)take((size_t)T_TOK * DIM * 2);
  unsigned short* w1b  = (unsigned short*)take(WREXP * 2);
  unsigned short* w2b  = (unsigned short*)take(WREXP * 2);
  unsigned short* w3b  = (unsigned short*)take(WREXP * 2);
  unsigned short* sw1b = (unsigned short*)take(WRSH * 2);
  unsigned short* sw2b = (unsigned short*)take(WRSH * 2);
  unsigned short* sw3b = (unsigned short*)take(WRSH * 2);
  unsigned short* hsh  = (unsigned short*)take((size_t)T_TOK * SIDIM * 2);
  unsigned short* hrt  = (unsigned short*)take((size_t)2 * T_TOK * IDIM * 2);
  int*   topi   = (int*)take((size_t)T_TOK * 2 * 4);
  float* topw   = (float*)take((size_t)T_TOK * 2 * 4);
  int*   counts = (int*)take(256);
  int*   segOff = (int*)take(256);
  int*   cursor = (int*)take(256);
  int*   rowsA  = (int*)take((size_t)2 * T_TOK * 4);
  float* rowWA  = (float*)take((size_t)2 * T_TOK * 4);

  // 1) fp32 -> bf16 streaming conversions (x + all weights)
  auto cvt = [&](const float* s, unsigned short* d, size_t n) {
    int n4 = (int)(n / 4);
    cvt_bf16_kernel<<<(n4 + 255) / 256, 256, 0, stream>>>(s, d, n4);
  };
  cvt(x,   xb,   (size_t)T_TOK * DIM);
  cvt(w1,  w1b,  WREXP);
  cvt(w2,  w2b,  WREXP);
  cvt(w3,  w3b,  WREXP);
  cvt(sw1, sw1b, WRSH);
  cvt(sw2, sw2b, WRSH);
  cvt(sw3, sw3b, WRSH);
  // 2) gating + routing tables
  zero_counts_kernel<<<1, 32, 0, stream>>>(counts);
  gate_topk_kernel<<<T_TOK / 8, 256, 0, stream>>>(x, gw, topi, topw, counts);
  prefix8_kernel<<<1, 32, 0, stream>>>(counts, segOff, cursor);
  scatter_rows_kernel<<<T_TOK / 256, 256, 0, stream>>>(topi, topw, cursor, rowsA, rowWA);
  // 3) shared expert: up then down (down initializes d_out)
  dim3 gsu(SIDIM / TILE_N, T_TOK / TILE_M, 1);
  up_swiglu_kernel<false><<<gsu, 256, 0, stream>>>(xb, sw1b, sw3b, hsh, nullptr, nullptr,
                                                   SIDIM, DIM, T_TOK);
  dim3 gsd(DIM / TILE_N, T_TOK / TILE_M, 1);
  down_proj_kernel<false><<<gsd, 256, 0, stream>>>(hsh, sw2b, out, nullptr, nullptr, nullptr,
                                                   DIM, SIDIM, T_TOK);
  // 4) routed experts: grouped GEMMs over gathered segments, atomic combine
  dim3 gru(IDIM / TILE_N, T_TOK / TILE_M, NEXP);
  up_swiglu_kernel<true><<<gru, 256, 0, stream>>>(xb, w1b, w3b, hrt, rowsA, segOff,
                                                  IDIM, DIM, 0);
  dim3 grd(DIM / TILE_N, T_TOK / TILE_M, NEXP);
  down_proj_kernel<true><<<grd, 256, 0, stream>>>(hrt, w2b, out, rowsA, rowWA, segOff,
                                                  DIM, IDIM, 0);
}
